// DCRNNModel_classification_57354993271297
// MI455X (gfx1250) — matrix operand order repaired and checked by reading, not compiled, not run on hardware
//
#include <hip/hip_runtime.h>
#include <hip/hip_bf16.h>
#include <math.h>

// ---------------------------------------------------------------------------
// DCRNN classification, persistent-per-batch DCGRU scan for gfx1250 (MI455X).
//   * one workgroup (256 thr = 8 wave32) per batch element, T=128 step loop
//   * all GEMMs via v_wmma_f32_16x16x32_bf16 (bf16 A/B, f32 accum)
//   * recurrent state h0/h1 fp32 resident in LDS for the whole sequence
//   * weights pre-transposed to bf16 (K-contiguous) in d_ws, streamed from L2
//   * input slab for step t+1 DMA'd by the Tensor Data Mover into LDS while
//     step t computes (tensor_load_to_lds + s_wait_tensorcnt pipeline)
// ---------------------------------------------------------------------------

typedef __bf16 bf16;
typedef __attribute__((ext_vector_type(16))) __bf16 bfx16;
typedef __attribute__((ext_vector_type(8)))  __bf16 bfx8;
typedef __attribute__((ext_vector_type(8)))  float  fx8;
typedef __attribute__((ext_vector_type(4)))  unsigned int u32x4;
typedef __attribute__((ext_vector_type(8)))  int          i32x8;
typedef __attribute__((ext_vector_type(4)))  int          i32x4;

#define T_STEPS 128
#define NN      64      // nodes
#define DD      64      // input features
#define HH      64      // hidden
#define CW      128     // concat width D+H
#define KC      640     // 5 * CW
#define NBATCH  32
#define NC      4
#define RS_XCAT 648     // xcat row stride (elems), 16B-aligned, bank-staggered
#define RS_T    72      // transpose-buffer / support row stride (elems)

// d_ws layout (bytes): bf16 transposed weights, [n][k] with k contiguous
#define WS_WGT0 0
#define WS_WCT0 (WS_WGT0 + 128 * KC * 2)
#define WS_WGT1 (WS_WCT0 +  64 * KC * 2)
#define WS_WCT1 (WS_WGT1 + 128 * KC * 2)

union BFrag { bfx16 v; bfx8 h[2]; };

__device__ __forceinline__ fx8 wmma_bf16(const BFrag& a, const BFrag& b, fx8 c) {
  // D = A(16x32 bf16) * B(32x16 bf16) + C(f32)
  return __builtin_amdgcn_wmma_f32_16x16x32_bf16(false, a.v, false, b.v,
                                                 (short)0, c, false, false);
}

// A operand, 16x32 tile from node-major storage (row = M, contiguous K).
// ISA 7.12.2 layout: lanes 0-15 rows M0-15 {K 0-7 | 16-23}, lanes 16-31 {8-15 | 24-31}
__device__ __forceinline__ BFrag load_a_frag(const bf16* base, int rs, int m0,
                                             int k0, int lane) {
  int row = m0 + (lane & 15);
  int kg  = lane >> 4;
  const bf16* p = base + row * rs + k0 + kg * 8;
  BFrag f;
  f.h[0] = *(const bfx8*)(p);
  f.h[1] = *(const bfx8*)(p + 16);
  return f;
}

// B operand, 32x16 tile from transposed (N-major, K-contiguous) storage:
// lane n (0-15): col N=n, K {0-7 | 16-23} ; lanes 16-31: same cols, K {8-15 | 24-31}
__device__ __forceinline__ BFrag load_b_frag(const bf16* baseT, int rs, int n0,
                                             int k0, int lane) {
  int col = n0 + (lane & 15);
  int kg  = lane >> 4;
  const bf16* p = baseT + col * rs + k0 + kg * 8;
  BFrag f;
  f.h[0] = *(const bfx8*)(p);
  f.h[1] = *(const bfx8*)(p + 16);
  return f;
}

__device__ __forceinline__ unsigned short f2bf_bits(float x) {
  unsigned u = __float_as_uint(x);
  unsigned r = u + 0x7FFFu + ((u >> 16) & 1u);   // round-to-nearest-even
  return (unsigned short)(r >> 16);
}

// ---------------------------------------------------------------------------
// Prep: transpose Wg/Wc to bf16 [n][k] (k contiguous) so WMMA B-fragments are
// two contiguous b128 loads per lane straight out of L2.
// ---------------------------------------------------------------------------
__global__ __launch_bounds__(256)
void prep_weights_kernel(const float* __restrict__ Wg0, const float* __restrict__ Wc0,
                         const float* __restrict__ Wg1, const float* __restrict__ Wc1,
                         unsigned short* __restrict__ wgT0, unsigned short* __restrict__ wcT0,
                         unsigned short* __restrict__ wgT1, unsigned short* __restrict__ wcT1) {
  int i = blockIdx.x * blockDim.x + threadIdx.x;
  if (i < 128 * KC) {                       // Wg: (640,128) -> (128,640)
    int n = i / KC, k = i - n * KC;
    wgT0[i] = f2bf_bits(Wg0[k * 128 + n]);
    wgT1[i] = f2bf_bits(Wg1[k * 128 + n]);
  }
  if (i < 64 * KC) {                        // Wc: (640,64) -> (64,640)
    int n = i / KC, k = i - n * KC;
    wcT0[i] = f2bf_bits(Wc0[k * 64 + n]);
    wcT1[i] = f2bf_bits(Wc1[k * 64 + n]);
  }
}

// ---------------------------------------------------------------------------
// Persistent DCGRU kernel: grid = 32 (one batch element / WGP), block = 256.
// ---------------------------------------------------------------------------
__global__ __launch_bounds__(256)
void dcrnn_persistent_kernel(const float* __restrict__ xin,   // (B,T,N,D)
                             const int*   __restrict__ slen,  // (B,)
                             const float* __restrict__ S0f,
                             const float* __restrict__ S1f,
                             const unsigned short* __restrict__ wgT0u,
                             const unsigned short* __restrict__ wcT0u,
                             const unsigned short* __restrict__ wgT1u,
                             const unsigned short* __restrict__ wcT1u,
                             const float* __restrict__ bg0,
                             const float* __restrict__ bc0,
                             const float* __restrict__ bg1,
                             const float* __restrict__ bc1,
                             const float* __restrict__ Wp,
                             const float* __restrict__ bp,
                             float* __restrict__ out) {
  __shared__ __align__(16) bf16  s_xcat[NN * RS_XCAT];   // 81 KB: [x | x1s0 | x2s0 | x1s1 | x2s1]
  __shared__ __align__(16) bf16  s_bT[CW * RS_T];        // 18 KB: K-contiguous transpose of B source
  __shared__ __align__(16) bf16  s_S[2][NN * RS_T];      // 18 KB: supports, bf16 node-major
  __shared__ __align__(16) float s_h[2][NN * HH];        // 32 KB: recurrent state, fp32
  __shared__ __align__(16) bf16  s_u[NN * HH];           //  8 KB: update gate
  __shared__ __align__(16) float s_stage[NN * DD];       // 16 KB: TDM staging for x_t (fp32)

  const int tid  = threadIdx.x;
  const int wave = tid >> 5;
  const int lane = tid & 31;
  const int b    = blockIdx.x;

  const bf16* wgT[2] = { (const bf16*)wgT0u, (const bf16*)wgT1u };
  const bf16* wcT[2] = { (const bf16*)wcT0u, (const bf16*)wcT1u };
  const float* bg[2] = { bg0, bg1 };
  const float* bc[2] = { bc0, bc1 };

  // ---- TDM: async-DMA one (64x64 fp32) input slab into s_stage -----------
  // D# per CDNA5 ISA: group0 = {count, lds_addr,
  // global_addr, type=2}; group1 = {data_size=4B, dim0=tile0=4096, tile1=1,
  // stride0=4096}; groups 2/3 disabled (<=2D tensor) -> zero-filled.
  // One wave issues; EXEC is ignored by TDM; completion via TENSORcnt.
  auto issue_tdm = [&](int tt) {
    unsigned long long ga =
        (unsigned long long)(uintptr_t)(xin + (((size_t)b * T_STEPS + tt) * NN * DD));
    unsigned int lds_off = (unsigned int)(uintptr_t)(&s_stage[0]);  // addr[31:0] = LDS offset
    u32x4 g0;
    g0[0] = 1u;                                             // count=1 (valid user D#)
    g0[1] = lds_off;                                        // lds_addr (bytes)
    g0[2] = (unsigned int)(ga & 0xFFFFFFFFu);               // global_addr[31:0]
    g0[3] = (unsigned int)((ga >> 32) & 0x01FFFFFFu)        // global_addr[56:32]
          | (2u << 30);                                     // type=2 ("image")
    i32x8 g1;
    g1[0] = (int)(2u << 16);                                // data_size=4B; mask=0
    g1[1] = (int)(4096u << 16);                             // tensor_dim0[15:0]=4096
    g1[2] = (int)(1u << 16);                                // dim0 hi=0; tensor_dim1=1
    g1[3] = (int)(4096u << 16);                             // tile_dim0=4096
    g1[4] = 1;                                              // tile_dim1=1, tile_dim2=0
    g1[5] = 4096;                                           // tensor_dim0_stride=4096
    g1[6] = 0;
    g1[7] = 0;
    i32x4 z4 = {0, 0, 0, 0};
    i32x8 z8 = {0, 0, 0, 0, 0, 0, 0, 0};
    __builtin_amdgcn_tensor_load_to_lds(g0, g1, z4, z4, z8, 0);
  };

  // ---- one-time block init: supports -> bf16 LDS, zero state -------------
  for (int i = tid; i < NN * NN; i += 256) {
    int m = i >> 6, n = i & 63;
    s_S[0][m * RS_T + n] = (bf16)S0f[i];
    s_S[1][m * RS_T + n] = (bf16)S1f[i];
    s_h[0][i] = 0.f;
    s_h[1][i] = 0.f;
  }
  if (wave == 0) issue_tdm(0);          // prime the pipeline with slab t=0
  __syncthreads();

  int sl = slen[b] - 1;
  const int idx = sl < 0 ? 0 : (sl > T_STEPS - 1 ? T_STEPS - 1 : sl);

  float lastReg[16];
#pragma unroll
  for (int k = 0; k < 16; ++k) lastReg[k] = 0.f;

  // transpose region 'reg' of xcat (64 x 128, node-major) into s_bT (K-contig)
  auto buildT = [&](int reg) {
    const bf16* src = s_xcat + reg * CW;
    for (int i = tid; i < NN * CW; i += 256) {
      int r = i >> 7, c = i & 127;
      s_bT[c * RS_T + r] = src[r * RS_XCAT + c];
    }
  };

  // x_out(64x128) = S_s @ x ; cheb: x_out = 2*S_s@x - x0(region 0)
  auto diff_gemm = [&](const bf16* A, int dstReg, bool cheb) {
    const int mT = wave >> 1;
    const int nB = (wave & 1) * 64;
    fx8 acc[4] = {};
#pragma unroll
    for (int k0 = 0; k0 < NN; k0 += 32) {
      BFrag a = load_a_frag(A, RS_T, mT * 16, k0, lane);
#pragma unroll
      for (int j = 0; j < 4; ++j) {
        BFrag bb = load_b_frag(s_bT, RS_T, nB + j * 16, k0, lane);
        acc[j] = wmma_bf16(a, bb, acc[j]);
      }
    }
    const int rbase = mT * 16 + (lane >> 4) * 8;
#pragma unroll
    for (int j = 0; j < 4; ++j) {
      const int col = nB + j * 16 + (lane & 15);
#pragma unroll
      for (int r = 0; r < 8; ++r) {
        const int row = rbase + r;
        float v = acc[j][r];
        if (cheb) v = 2.f * v - (float)s_xcat[row * RS_XCAT + col];
        s_xcat[row * RS_XCAT + dstReg * CW + col] = (bf16)v;
      }
    }
  };

#pragma unroll 1
  for (int t = 0; t < T_STEPS; ++t) {
#pragma unroll 1
    for (int l = 0; l < 2; ++l) {
      // ---- build x = concat([xt, h_l]) into xcat region 0 ----------------
      if (l == 0) {
        if (wave == 0) __builtin_amdgcn_s_wait_tensorcnt(0);  // slab t landed
        __syncthreads();
        for (int i = tid; i < NN * DD; i += 256) {
          int r = i >> 6, c = i & 63;
          s_xcat[r * RS_XCAT + c] = (bf16)s_stage[i];
        }
      } else {
        for (int i = tid; i < NN * DD; i += 256) {
          int r = i >> 6, c = i & 63;
          s_xcat[r * RS_XCAT + c] = (bf16)s_h[0][i];
        }
      }
      for (int i = tid; i < NN * HH; i += 256) {
        int r = i >> 6, c = i & 63;
        s_xcat[r * RS_XCAT + DD + c] = (bf16)s_h[l][i];
      }
      __syncthreads();
      // staging consumed by everyone -> kick off DMA for the next slab; it
      // overlaps with all of this step's diffusion / gate / candidate GEMMs
      if (l == 0 && wave == 0 && t + 1 < T_STEPS) issue_tdm(t + 1);

#pragma unroll 1
      for (int pass = 0; pass < 2; ++pass) {       // 0: gate (r,u)  1: candidate
        // ---- diffusion: fill regions 1..4 --------------------------------
#pragma unroll 1
        for (int s = 0; s < 2; ++s) {
          buildT(0);            __syncthreads();   // x^T
          diff_gemm(s_S[s], 1 + 2 * s, false);
          __syncthreads();
          buildT(1 + 2 * s);    __syncthreads();   // x1^T
          diff_gemm(s_S[s], 2 + 2 * s, true);
          __syncthreads();
        }

        if (pass == 0) {
          // ---- ru = sigmoid(xcat(64x640) @ Wg(640x128) + bg) -------------
          const int mT = wave >> 1;
          const int nB = (wave & 1) * 64;
          fx8 acc[4] = {};
          for (int k0 = 0; k0 < KC; k0 += 32) {
            BFrag a = load_a_frag(s_xcat, RS_XCAT, mT * 16, k0, lane);
#pragma unroll
            for (int j = 0; j < 4; ++j) {
              BFrag bb = load_b_frag(wgT[l], KC, nB + j * 16, k0, lane);
              acc[j] = wmma_bf16(a, bb, acc[j]);
            }
          }
          __syncthreads();  // all waves done reading xcat before r*h overwrite
          const int rbase = mT * 16 + (lane >> 4) * 8;
#pragma unroll
          for (int j = 0; j < 4; ++j) {
            const int col = nB + j * 16 + (lane & 15);
            const float bgv = bg[l][col];
#pragma unroll
            for (int r = 0; r < 8; ++r) {
              const int row = rbase + r;
              float g = 1.f / (1.f + __expf(-(acc[j][r] + bgv)));
              if (col < HH) {   // reset gate -> overwrite h-part with r*h
                float rh = g * s_h[l][row * HH + col];
                s_xcat[row * RS_XCAT + DD + col] = (bf16)rh;
              } else {          // update gate
                s_u[row * HH + (col - HH)] = (bf16)g;
              }
            }
          }
          __syncthreads();
        } else {
          // ---- c = tanh(xcat2 @ Wc(640x64) + bc); h = u*h + (1-u)*c ------
          const int mT = wave >> 1;
          const int nB = (wave & 1) * 32;
          fx8 acc[2] = {};
          for (int k0 = 0; k0 < KC; k0 += 32) {
            BFrag a = load_a_frag(s_xcat, RS_XCAT, mT * 16, k0, lane);
#pragma unroll
            for (int j = 0; j < 2; ++j) {
              BFrag bb = load_b_frag(wcT[l], KC, nB + j * 16, k0, lane);
              acc[j] = wmma_bf16(a, bb, acc[j]);
            }
          }
          __syncthreads();
          const int rbase = mT * 16 + (lane >> 4) * 8;
#pragma unroll
          for (int j = 0; j < 2; ++j) {
            const int col = nB + j * 16 + (lane & 15);
            const float bcv = bc[l][col];
#pragma unroll
            for (int r = 0; r < 8; ++r) {
              const int row = rbase + r;
              float c  = tanhf(acc[j][r] + bcv);
              float uu = (float)s_u[row * HH + col];
              float hp = s_h[l][row * HH + col];
              s_h[l][row * HH + col] = uu * hp + (1.f - uu) * c;
            }
          }
          __syncthreads();
        }
      } // pass
    } // layer

    if (t == idx) {   // snapshot out1[idx] into registers (16 floats/thread)
#pragma unroll
      for (int k = 0; k < 16; ++k) lastReg[k] = s_h[1][tid + (k << 8)];
    }
  } // t

  // ---- head: relu(last) @ Wp + bp, max over nodes --------------------------
  __syncthreads();
  float* s_last = (float*)s_xcat;   // xcat is dead; reuse as fp32 scratch
  float* s_red  = s_stage;          // staging is dead; reuse for reduction
#pragma unroll
  for (int k = 0; k < 16; ++k) s_last[tid + (k << 8)] = lastReg[k];
  __syncthreads();

  if (tid < NN) {
    float a[NC];
#pragma unroll
    for (int c = 0; c < NC; ++c) a[c] = bp[c];
    for (int f = 0; f < HH; ++f) {
      float v = s_last[tid * HH + f];
      v = v > 0.f ? v : 0.f;
#pragma unroll
      for (int c = 0; c < NC; ++c) a[c] += v * Wp[f * NC + c];
    }
#pragma unroll
    for (int c = 0; c < NC; ++c) s_red[tid * NC + c] = a[c];
  }
  __syncthreads();
  if (tid < NC) {
    float m = -INFINITY;
    for (int n = 0; n < NN; ++n) m = fmaxf(m, s_red[n * NC + tid]);
    out[b * NC + tid] = m;
  }
}

// ---------------------------------------------------------------------------
extern "C" void kernel_launch(void* const* d_in, const int* in_sizes, int n_in,
                              void* d_out, int out_size, void* d_ws, size_t ws_size,
                              hipStream_t stream) {
  (void)in_sizes; (void)n_in; (void)out_size; (void)ws_size;

  const float* xin  = (const float*)d_in[0];
  const int*   slen = (const int*)  d_in[1];
  const float* S0   = (const float*)d_in[2];
  const float* S1   = (const float*)d_in[3];
  const float* Wg0  = (const float*)d_in[4];
  const float* bg0  = (const float*)d_in[5];
  const float* Wc0  = (const float*)d_in[6];
  const float* bc0  = (const float*)d_in[7];
  const float* Wg1  = (const float*)d_in[8];
  const float* bg1  = (const float*)d_in[9];
  const float* Wc1  = (const float*)d_in[10];
  const float* bc1  = (const float*)d_in[11];
  const float* Wp   = (const float*)d_in[12];
  const float* bp   = (const float*)d_in[13];
  float* out = (float*)d_out;

  char* ws = (char*)d_ws;
  unsigned short* wgT0 = (unsigned short*)(ws + WS_WGT0);
  unsigned short* wcT0 = (unsigned short*)(ws + WS_WCT0);
  unsigned short* wgT1 = (unsigned short*)(ws + WS_WGT1);
  unsigned short* wcT1 = (unsigned short*)(ws + WS_WCT1);

  prep_weights_kernel<<<(128 * KC + 255) / 256, 256, 0, stream>>>(
      Wg0, Wc0, Wg1, Wc1, wgT0, wcT0, wgT1, wcT1);

  dcrnn_persistent_kernel<<<NBATCH, 256, 0, stream>>>(
      xin, slen, S0, S1, wgT0, wcT0, wgT1, wcT1,
      bg0, bc0, bg1, bc1, Wp, bp, out);
}
